// AudioMamba_77266461655130
// MI455X (gfx1250) — matrix operand order repaired and checked by reading, not compiled, not run on hardware
//
#include <hip/hip_runtime.h>
#include <math.h>

#define DEPTH    24
#define D_MODEL  192
#define D_INNER  384
#define D_STATE  16
#define DT_RANK  12
#define IMG_H    32
#define IMG_W    3200
#define PATCH    16
#define NPATCH   400      // (3200/16)*(32/16)
#define SEQLEN   401      // NPATCH + 1 (cls)
#define NCLS     22
#define BATCH    4
#define BLTOK    (BATCH*SEQLEN)            // 1604 tokens
#define XPROJ_N  (DT_RANK + 2*D_STATE)     // 44

typedef __attribute__((ext_vector_type(16))) _Float16 v16h;
typedef __attribute__((ext_vector_type(8)))  float    v8f;

__device__ __forceinline__ float silu_f(float x) { return x * (1.f / (1.f + __expf(-x))); }

__device__ __forceinline__ void cvt8(_Float16* dst, const float4 lo, const float4 hi)
{
    dst[0] = (_Float16)lo.x; dst[1] = (_Float16)lo.y;
    dst[2] = (_Float16)lo.z; dst[3] = (_Float16)lo.w;
    dst[4] = (_Float16)hi.x; dst[5] = (_Float16)hi.y;
    dst[6] = (_Float16)hi.z; dst[7] = (_Float16)hi.w;
}

__device__ __forceinline__ float4 add4(const float4 a, const float4 b)
{
    return make_float4(a.x + b.x, a.y + b.y, a.z + b.z, a.w + b.w);
}

// ---------------------------------------------------------------------------
// GEMM:  C[M,N] = (A [+ A2])[M,K] @ W[N,K]^T (+ bias)
//  * K must be a multiple of 32 (true for all uses: 192, 256, 384).
//  * One wave computes a 64(M)x16(N) strip: 4 accumulators reuse one B frag.
//  * Edge rows/cols are CLAMPED for loads (always in-bounds, no exec churn),
//    guarded only at the store.  All loads are float4 -> global_load_b128.
//  * Per K-step, ALL 20 (or 36) b128 loads are issued into distinct registers
//    before any convert, so they clause together and overlap (one wait,
//    then straight-line cvt + 4 back-to-back WMMAs).
//  * blockDim = 128 (4 waves -> 64 N), grid = (ceil(N/64), ceil(M/64)).
// ---------------------------------------------------------------------------
template <bool HAS_A2>
__global__ void am_gemm_wmma(const float* __restrict__ A, const float* __restrict__ A2,
                             const float* __restrict__ W, const float* __restrict__ Bias,
                             float* __restrict__ C, int M, int N, int K)
{
    const int lane = threadIdx.x & 31;
    const int wave = threadIdx.x >> 5;
    const int n0 = (blockIdx.x * (blockDim.x >> 5) + wave) * 16;
    const int m_base = blockIdx.y * 64;
    const int row  = lane & 15;
    const int half = lane >> 4;

    // Clamped row pointers: every load below is unconditionally in-bounds.
    const int bn = min(n0 + row, N - 1);
    const float4* __restrict__ Wrow4 = (const float4*)(W + (size_t)bn * K);

    const float4* __restrict__ Arow4[4];
    const float4* __restrict__ A2row4[4];
#pragma unroll
    for (int s = 0; s < 4; ++s) {
        const int am = min(m_base + s * 16 + row, M - 1);
        Arow4[s] = (const float4*)(A + (size_t)am * K);
        if (HAS_A2) A2row4[s] = (const float4*)(A2 + (size_t)am * K);
    }

    v8f acc[4] = {{}, {}, {}, {}};
    const int K4 = K >> 2;

    for (int k4 = 0; k4 < K4; k4 += 8) {            // 32 K-values per iteration
        const int kb4  = k4 + half * 4;             // B: 16 contiguous floats
        const int ka4l = k4 + half * 4;             // A: K = k0 + half*16 + 0..7
        const int ka4h = k4 + half * 4 + 2;         // A: K = k0 + half*16 + 8..15

        // ---- Issue every load of this K-step first (distinct registers) ----
        float4 br[4];
#pragma unroll
        for (int j = 0; j < 4; ++j) br[j] = Wrow4[kb4 + j];

        float4 ar[4][4];
#pragma unroll
        for (int s = 0; s < 4; ++s) {
            ar[s][0] = Arow4[s][ka4l + 0];
            ar[s][1] = Arow4[s][ka4l + 1];
            ar[s][2] = Arow4[s][ka4h + 0];
            ar[s][3] = Arow4[s][ka4h + 1];
        }
        if (HAS_A2) {
            float4 ar2[4][4];
#pragma unroll
            for (int s = 0; s < 4; ++s) {
                ar2[s][0] = A2row4[s][ka4l + 0];
                ar2[s][1] = A2row4[s][ka4l + 1];
                ar2[s][2] = A2row4[s][ka4h + 0];
                ar2[s][3] = A2row4[s][ka4h + 1];
            }
#pragma unroll
            for (int s = 0; s < 4; ++s)
#pragma unroll
                for (int j = 0; j < 4; ++j) ar[s][j] = add4(ar[s][j], ar2[s][j]);
        }
        __builtin_prefetch((const float*)(Wrow4 + kb4 + 8), 0, 1);

        // ---- Convert + 4 WMMAs, no loads interleaved ----
        union { v16h v; _Float16 e[16]; } bf;
        cvt8(bf.e,     br[0], br[1]);
        cvt8(bf.e + 8, br[2], br[3]);
#pragma unroll
        for (int s = 0; s < 4; ++s) {
            union { v16h v; _Float16 e[16]; } af;
            cvt8(af.e,     ar[s][0], ar[s][1]);
            cvt8(af.e + 8, ar[s][2], ar[s][3]);
            acc[s] = __builtin_amdgcn_wmma_f32_16x16x32_f16(
                false, af.v, false, bf.v, (short)0, acc[s], false, false);
        }
    }

    const int n = n0 + row;
    if (n < N) {
        const float b = Bias ? Bias[n] : 0.f;
#pragma unroll
        for (int s = 0; s < 4; ++s) {
#pragma unroll
            for (int r = 0; r < 8; ++r) {
                int m = m_base + s * 16 + half * 8 + r;
                if (m < M) C[(size_t)m * N + n] = acc[s][r] + b;
            }
        }
    }
}

// ---------------------------------------------------------------------------
// im2col: extract 16x16 patches of x (B,32,3200) -> patches[B*400, 256]
// ---------------------------------------------------------------------------
__global__ void am_im2col(const float* __restrict__ x, float* __restrict__ patches)
{
    size_t idx = (size_t)blockIdx.x * blockDim.x + threadIdx.x;
    const size_t total = (size_t)BATCH * NPATCH * 256;
    if (idx >= total) return;
    int e = idx % 256;
    int p = (idx / 256) % NPATCH;
    int b = idx / (256 * NPATCH);
    int i = e >> 4, j = e & 15;
    int ph = p / 200, pw = p % 200;
    patches[idx] = x[(size_t)b * IMG_H * IMG_W + (size_t)(ph * 16 + i) * IMG_W + (pw * 16 + j)];
}

// tok = concat(cls, patch_out) + pos_embed ; hidden = residual = tok
__global__ void am_assemble_tokens(const float* __restrict__ patch_out,
                                   const float* __restrict__ cls,
                                   const float* __restrict__ pos,
                                   float* __restrict__ hidden, float* __restrict__ residual)
{
    size_t idx = (size_t)blockIdx.x * blockDim.x + threadIdx.x;
    const size_t total = (size_t)BLTOK * D_MODEL;
    if (idx >= total) return;
    int c = idx % D_MODEL;
    int t = (idx / D_MODEL) % SEQLEN;
    int b = idx / ((size_t)D_MODEL * SEQLEN);
    float v = (t == 0) ? cls[c] : patch_out[((size_t)(b * NPATCH + t - 1)) * D_MODEL + c];
    v += pos[(size_t)t * D_MODEL + c];
    hidden[idx] = v;
    residual[idx] = v;
}

// ---------------------------------------------------------------------------
// residual += hidden ; out = rmsnorm(residual) * w.  One block (192 thr) / token.
// ---------------------------------------------------------------------------
__global__ void am_add_rmsnorm(float* __restrict__ residual, const float* __restrict__ hidden,
                               const float* __restrict__ w, float* __restrict__ out)
{
    __shared__ float s[D_MODEL];
    const int t = blockIdx.x, c = threadIdx.x;
    const size_t i = (size_t)t * D_MODEL + c;
    float r = residual[i] + hidden[i];
    residual[i] = r;
    s[c] = r * r;
    __syncthreads();
    if (c < 64) s[c] += s[c + 128];
    __syncthreads();
    if (c < 64) s[c] += s[c + 64];
    __syncthreads();
    for (int st = 32; st > 0; st >>= 1) { if (c < st) s[c] += s[c + st]; __syncthreads(); }
    float ms = s[0] / (float)D_MODEL;
    out[i] = r * rsqrtf(ms + 1e-5f) * w[c];
}

// ---------------------------------------------------------------------------
// Causal depthwise conv (K=4) + SiLU over xx = xz[..., :384].
// dir 0: natural time -> xxf ; dir 1: reversed time -> xxb (kept reversed).
// ---------------------------------------------------------------------------
__global__ void am_conv_silu(const float* __restrict__ xz, const float* __restrict__ cw,
                             const float* __restrict__ cb,
                             float* __restrict__ xxf, float* __restrict__ xxb)
{
    size_t idx = (size_t)blockIdx.x * blockDim.x + threadIdx.x;
    const size_t total = (size_t)2 * BLTOK * D_INNER;
    if (idx >= total) return;
    int d = idx % D_INNER;
    int l = (idx / D_INNER) % SEQLEN;
    int b = (idx / ((size_t)D_INNER * SEQLEN)) % BATCH;
    int dir = idx / ((size_t)D_INNER * SEQLEN * BATCH);
    float acc = cb[d];
#pragma unroll
    for (int k = 0; k < 4; ++k) {
        int li = l - 3 + k;
        if (li < 0) continue;
        int src_l = (dir == 0) ? li : (SEQLEN - 1 - li);
        acc += cw[d * 4 + k] * xz[((size_t)(b * SEQLEN + src_l)) * (2 * D_INNER) + d];
    }
    float y = silu_f(acc);
    (dir == 0 ? xxf : xxb)[((size_t)(b * SEQLEN + l)) * D_INNER + d] = y;
}

// dt = softplus(dbl[:, :12] @ dtw^T + dtb)  (K=12 -> trivial VALU dot)
__global__ void am_dt_softplus(const float* __restrict__ dblf, const float* __restrict__ dblb,
                               const float* __restrict__ dtw, const float* __restrict__ dtb,
                               float* __restrict__ dtf, float* __restrict__ dtbo)
{
    size_t idx = (size_t)blockIdx.x * blockDim.x + threadIdx.x;
    const size_t total = (size_t)2 * BLTOK * D_INNER;
    if (idx >= total) return;
    int d = idx % D_INNER;
    size_t m = (idx / D_INNER) % BLTOK;
    int dir = idx / ((size_t)D_INNER * BLTOK);
    const float* row = (dir ? dblb : dblf) + m * XPROJ_N;
    float acc = dtb[d];
#pragma unroll
    for (int r = 0; r < DT_RANK; ++r) acc += row[r] * dtw[d * DT_RANK + r];
    float sp = (acc > 20.f) ? acc : log1pf(__expf(acc));
    (dir ? dtbo : dtf)[m * D_INNER + d] = sp;
}

// ---------------------------------------------------------------------------
// Selective scan: one thread per (dir, batch, channel) = 3072 recurrences,
// 16-state recurrence in registers, gated by silu(z), backward dir scattered
// back to natural time so out_proj can fuse y_f + y_b.
// ---------------------------------------------------------------------------
__global__ void am_selective_scan(const float* __restrict__ xxf, const float* __restrict__ xxb,
                                  const float* __restrict__ dtf, const float* __restrict__ dtb,
                                  const float* __restrict__ dblf, const float* __restrict__ dblb,
                                  const float* __restrict__ Alog, const float* __restrict__ Ablog,
                                  const float* __restrict__ Dsk, const float* __restrict__ xz,
                                  float* __restrict__ yf, float* __restrict__ yb)
{
    int idx = blockIdx.x * blockDim.x + threadIdx.x;
    if (idx >= 2 * BATCH * D_INNER) return;
    int d = idx % D_INNER;
    int b = (idx / D_INNER) % BATCH;
    int dir = idx / (D_INNER * BATCH);

    const float* xx  = dir ? xxb : xxf;
    const float* dt  = dir ? dtb : dtf;
    const float* dbl = dir ? dblb : dblf;
    const float* Al  = dir ? Ablog : Alog;
    float* yo        = dir ? yb : yf;

    float A[D_STATE], h[D_STATE];
#pragma unroll
    for (int n = 0; n < D_STATE; ++n) { A[n] = -__expf(Al[d * D_STATE + n]); h[n] = 0.f; }
    const float Dv = Dsk[d];

    for (int t = 0; t < SEQLEN; ++t) {
        const size_t tok = (size_t)(b * SEQLEN + t);
        const float xv  = xx[tok * D_INNER + d];
        const float dtv = dt[tok * D_INNER + d];
        const float* bc = dbl + tok * XPROJ_N;
        float y = 0.f;
#pragma unroll
        for (int n = 0; n < D_STATE; ++n) {
            float dA = __expf(dtv * A[n]);
            h[n] = dA * h[n] + dtv * xv * bc[DT_RANK + n];
            y += h[n] * bc[DT_RANK + D_STATE + n];
        }
        y += xv * Dv;
        int lz = dir ? (SEQLEN - 1 - t) : t;           // natural-time index
        float z = xz[((size_t)(b * SEQLEN + lz)) * (2 * D_INNER) + D_INNER + d];
        yo[((size_t)(b * SEQLEN + lz)) * D_INNER + d] = y * silu_f(z);
    }
}

// Final: rmsnorm(residual + hidden)[cls] @ head_w^T + head_b
__global__ void am_final_head(const float* __restrict__ residual, const float* __restrict__ hidden,
                              const float* __restrict__ nfw, const float* __restrict__ hw,
                              const float* __restrict__ hb, float* __restrict__ out)
{
    __shared__ float s[D_MODEL];
    __shared__ float v[D_MODEL];
    const int b = blockIdx.x, c = threadIdx.x;
    const size_t base = (size_t)b * SEQLEN * D_MODEL;  // cls token = position 0
    float r = residual[base + c] + hidden[base + c];
    s[c] = r * r;
    __syncthreads();
    if (c < 64) s[c] += s[c + 128];
    __syncthreads();
    if (c < 64) s[c] += s[c + 64];
    __syncthreads();
    for (int st = 32; st > 0; st >>= 1) { if (c < st) s[c] += s[c + st]; __syncthreads(); }
    float ms = s[0] / (float)D_MODEL;
    v[c] = r * rsqrtf(ms + 1e-5f) * nfw[c];
    __syncthreads();
    if (c < NCLS) {
        float acc = hb[c];
        for (int k = 0; k < D_MODEL; ++k) acc += v[k] * hw[c * D_MODEL + k];
        out[b * NCLS + c] = acc;
    }
}

// ---------------------------------------------------------------------------
extern "C" void kernel_launch(void* const* d_in, const int* in_sizes, int n_in,
                              void* d_out, int out_size, void* d_ws, size_t ws_size,
                              hipStream_t stream)
{
    const float* x         = (const float*)d_in[0];
    const float* pe_w      = (const float*)d_in[1];
    const float* pe_b      = (const float*)d_in[2];
    const float* cls_tok   = (const float*)d_in[3];
    const float* pos_emb   = (const float*)d_in[4];
    const float* norm_w    = (const float*)d_in[5];
    const float* in_proj_w = (const float*)d_in[6];
    const float* cw        = (const float*)d_in[7];
    const float* cb        = (const float*)d_in[8];
    const float* xproj_w   = (const float*)d_in[9];
    const float* dtproj_w  = (const float*)d_in[10];
    const float* dtproj_b  = (const float*)d_in[11];
    const float* A_log     = (const float*)d_in[12];
    const float* A_b_log   = (const float*)d_in[13];
    const float* Dp        = (const float*)d_in[14];
    const float* outproj_w = (const float*)d_in[15];
    const float* norm_f_w  = (const float*)d_in[16];
    const float* head_w    = (const float*)d_in[17];
    const float* head_b    = (const float*)d_in[18];
    float* out = (float*)d_out;

    float* wsf = (float*)d_ws;
    size_t off = 0;
    auto alloc = [&](size_t n) { float* p = wsf + off; off += n; return p; };

    float* hidden    = alloc((size_t)BLTOK * D_MODEL);
    float* residual  = alloc((size_t)BLTOK * D_MODEL);
    float* hnorm     = alloc((size_t)BLTOK * D_MODEL);
    float* xz        = alloc((size_t)BLTOK * 2 * D_INNER);
    float* xxf       = alloc((size_t)BLTOK * D_INNER);
    float* xxb       = alloc((size_t)BLTOK * D_INNER);
    float* dblf      = alloc((size_t)BLTOK * XPROJ_N);
    float* dblb      = alloc((size_t)BLTOK * XPROJ_N);
    float* dtf       = alloc((size_t)BLTOK * D_INNER);
    float* dtb2      = alloc((size_t)BLTOK * D_INNER);
    float* yf        = alloc((size_t)BLTOK * D_INNER);
    float* yb        = alloc((size_t)BLTOK * D_INNER);
    float* patches   = alloc((size_t)BATCH * NPATCH * 256);
    float* patch_out = alloc((size_t)BATCH * NPATCH * D_MODEL);
    (void)ws_size; (void)in_sizes; (void)n_in; (void)out_size;

    auto cd = [](int a, int b) { return (a + b - 1) / b; };

    // ---- Patch embedding as GEMM --------------------------------------
    {
        int tot = BATCH * NPATCH * 256;
        am_im2col<<<cd(tot, 256), 256, 0, stream>>>(x, patches);
        dim3 g(cd(D_MODEL, 64), cd(BATCH * NPATCH, 64));
        am_gemm_wmma<false><<<g, 128, 0, stream>>>(patches, nullptr, pe_w, pe_b, patch_out,
                                                   BATCH * NPATCH, D_MODEL, 256);
        int tot2 = BLTOK * D_MODEL;
        am_assemble_tokens<<<cd(tot2, 256), 256, 0, stream>>>(patch_out, cls_tok, pos_emb,
                                                              hidden, residual);
    }

    // ---- 24 Mamba layers ----------------------------------------------
    const int totEl = 2 * BLTOK * D_INNER;
    for (int l = 0; l < DEPTH; ++l) {
        am_add_rmsnorm<<<BLTOK, D_MODEL, 0, stream>>>(residual, hidden,
                                                      norm_w + (size_t)l * D_MODEL, hnorm);
        {   // in_proj: [BL,192] @ [768,192]^T -> xz
            dim3 g(cd(2 * D_INNER, 64), cd(BLTOK, 64));
            am_gemm_wmma<false><<<g, 128, 0, stream>>>(hnorm, nullptr,
                in_proj_w + (size_t)l * 2 * D_INNER * D_MODEL, nullptr, xz,
                BLTOK, 2 * D_INNER, D_MODEL);
        }
        am_conv_silu<<<cd(totEl, 256), 256, 0, stream>>>(xz,
            cw + (size_t)l * D_INNER * 4, cb + (size_t)l * D_INNER, xxf, xxb);
        {   // x_proj for both directions: [BL,384] @ [44,384]^T
            dim3 g(cd(XPROJ_N, 64), cd(BLTOK, 64));
            const float* xpw = xproj_w + (size_t)l * XPROJ_N * D_INNER;
            am_gemm_wmma<false><<<g, 128, 0, stream>>>(xxf, nullptr, xpw, nullptr, dblf,
                                                       BLTOK, XPROJ_N, D_INNER);
            am_gemm_wmma<false><<<g, 128, 0, stream>>>(xxb, nullptr, xpw, nullptr, dblb,
                                                       BLTOK, XPROJ_N, D_INNER);
        }
        am_dt_softplus<<<cd(totEl, 256), 256, 0, stream>>>(dblf, dblb,
            dtproj_w + (size_t)l * D_INNER * DT_RANK, dtproj_b + (size_t)l * D_INNER,
            dtf, dtb2);
        am_selective_scan<<<cd(2 * BATCH * D_INNER, 128), 128, 0, stream>>>(
            xxf, xxb, dtf, dtb2, dblf, dblb,
            A_log + (size_t)l * D_INNER * D_STATE, A_b_log + (size_t)l * D_INNER * D_STATE,
            Dp + (size_t)l * D_INNER, xz, yf, yb);
        {   // out_proj with fused (y_f + y_b): [BL,384] @ [192,384]^T -> hidden
            dim3 g(cd(D_MODEL, 64), cd(BLTOK, 64));
            am_gemm_wmma<true><<<g, 128, 0, stream>>>(yf, yb,
                outproj_w + (size_t)l * D_MODEL * D_INNER, nullptr, hidden,
                BLTOK, D_MODEL, D_INNER);
        }
    }

    // ---- Final norm + classification head -----------------------------
    am_final_head<<<BATCH, D_MODEL, 0, stream>>>(residual, hidden, norm_f_w,
                                                 head_w, head_b, out);
}
